// ResidualQuantizer_2095944040593
// MI455X (gfx1250) — matrix operand order repaired and checked by reading, not compiled, not run on hardware
//
#include <hip/hip_runtime.h>

typedef __attribute__((ext_vector_type(16))) _Float16 v16h;
typedef __attribute__((ext_vector_type(8)))  _Float16 v8h;
typedef __attribute__((ext_vector_type(8)))  float    v8f;

#define B_SIZE   262144
#define DIM      32
#define KCB      256
#define SK_ITERS 50
#define SK_EPS   0.003f
#define LOG_B    12.476649250079015f   /* ln(2^18) */
#define LOG_K    5.545177444479562f    /* ln(2^8)  */
#define LOSS_SCALE (1.25f / 8388608.0f) /* (1+beta)/(B*D) */
#define NEG_INF  (-3.4e38f)
#define POS_INF  ( 3.4e38f)

// ---------------------------------------------------------------------------
// A-operand swizzle: WMMA 16-bit A layout wants, per lane, K order
// {0..7,16..23} (lanes 0-15) / {8..15,24..31} (lanes 16-31). Store the
// residual f16 mirror with halves permuted so each lane loads ONE contiguous
// v16h.  pos(k) swaps bits 3 and 4 of k.
__device__ __forceinline__ int aswz(int k) {
    return (k & ~24) | ((k & 8) << 1) | ((k & 16) >> 1);
}

// Order-preserving float<->uint map for atomicMin/Max on f32 values.
__device__ __forceinline__ unsigned fordered(float f) {
    unsigned u = __float_as_uint(f);
    return (u & 0x80000000u) ? ~u : (u | 0x80000000u);
}
__device__ __forceinline__ float funordered(unsigned o) {
    unsigned u = (o & 0x80000000u) ? (o & 0x7fffffffu) : ~o;
    return __uint_as_float(u);
}

// ---------------------------------------------------------------------------
// Prep: residual(f32)=z, residual_h(f16,swizzled)=z, codebooks -> f16,
// zero/ident accumulators.
__global__ __launch_bounds__(256) void rq_prep_kernel(
    const float* __restrict__ z,
    const float* __restrict__ cb0f, const float* __restrict__ cb1f,
    const float* __restrict__ cb2f,
    float* __restrict__ residual, _Float16* __restrict__ residual_h,
    _Float16* __restrict__ cbh0, _Float16* __restrict__ cbh1,
    _Float16* __restrict__ cbh2,
    float* __restrict__ v, float* __restrict__ lossAcc,
    unsigned* __restrict__ dminO, unsigned* __restrict__ dmaxO,
    unsigned* __restrict__ bad)
{
    size_t idx = (size_t)blockIdx.x * 256 + threadIdx.x;
    if (idx < (size_t)B_SIZE * DIM) {
        float x = z[idx];
        residual[idx] = x;
        size_t row = idx / DIM;
        int   k    = (int)(idx % DIM);
        residual_h[row * DIM + aswz(k)] = (_Float16)x;
    }
    if (blockIdx.x == 0) {
        for (int i = threadIdx.x; i < KCB * DIM; i += 256) {
            cbh0[i] = (_Float16)cb0f[i];
            cbh1[i] = (_Float16)cb1f[i];
            cbh2[i] = (_Float16)cb2f[i];
        }
        if (threadIdx.x < KCB) v[threadIdx.x] = 0.0f;
        if (threadIdx.x == 0) {
            *lossAcc = 0.0f;
            *dminO   = 0xFFFFFFFFu;  // ordered-identity for min
            *dmaxO   = 0u;           // ordered-identity for max
            *bad     = 0u;
        }
    }
}

// ---------------------------------------------------------------------------
// Levels 0/1: WMMA distances -> argmin -> residual update + loss.
// Block = 256 threads = 8 waves; wave handles 16 rows x 256 codes.
__global__ __launch_bounds__(256) void rq_level_kernel(
    const _Float16* __restrict__ cbh, const float* __restrict__ cbf,
    float* __restrict__ residual, _Float16* __restrict__ residual_h,
    float* __restrict__ codes_out, float* __restrict__ lossAcc)
{
    __shared__ __align__(32) _Float16 s_cb[KCB * DIM];
    __shared__ float s_c2[KCB];
    __shared__ int   s_code[8][16];

    int tid = threadIdx.x;
    for (int i = tid; i < (KCB * DIM) / 8; i += 256)
        ((v8h*)s_cb)[i] = ((const v8h*)cbh)[i];
    __syncthreads();
    if (tid < KCB) {
        float s = 0.0f;
        for (int k = 0; k < DIM; ++k) {
            float x = (float)s_cb[tid * DIM + k];
            s += x * x;
        }
        s_c2[tid] = s;
    }
    __syncthreads();

    int wave = tid >> 5, lane = tid & 31;
    int l15 = lane & 15, hi = lane >> 4;
    int rowBase = (blockIdx.x * 8 + wave) * 16;

    // A operand: one contiguous 32B load per lane (pre-swizzled layout)
    v16h a = *(const v16h*)(residual_h + (size_t)(rowBase + l15) * DIM + hi * 16);

    // ||r||^2 from the same f16 values the WMMA consumes (consistent argmin)
    float r2p = 0.0f;
    for (int p = 0; p < 16; ++p) { float x = (float)a[p]; r2p += x * x; }
    r2p += __shfl_xor(r2p, 16, 32);
    float r2g[8];
    for (int g = 0; g < 8; ++g) r2g[g] = __shfl(r2p, g + 8 * hi, 32);

    float best[8]; int bidx[8];
    for (int g = 0; g < 8; ++g) { best[g] = POS_INF; bidx[g] = 0; }

    for (int t = 0; t < 16; ++t) {
        const _Float16* bp = s_cb + (size_t)(t * 16 + l15) * DIM + hi * 8;
        v8h b0 = *(const v8h*)bp, b1 = *(const v8h*)(bp + 16);
        v16h b = __builtin_shufflevector(b0, b1, 0,1,2,3,4,5,6,7,8,9,10,11,12,13,14,15);
        v8f c = {};
        c = __builtin_amdgcn_wmma_f32_16x16x32_f16(false, a, false, b,
                                                   (short)0, c, false, false);
        int col = t * 16 + l15;
        float c2v = s_c2[col];
        for (int g = 0; g < 8; ++g) {
            float d = r2g[g] + c2v - 2.0f * c[g];
            if (d < best[g]) { best[g] = d; bidx[g] = col; }
        }
    }
    // reduce over the 16 lanes of each half (C layout: halves = row groups)
    for (int g = 0; g < 8; ++g) {
        float bv = best[g]; int bi = bidx[g];
        for (int m = 8; m >= 1; m >>= 1) {
            float ov = __shfl_xor(bv, m, 16);
            int   oi = __shfl_xor(bi, m, 16);
            if (ov < bv || (ov == bv && oi < bi)) { bv = ov; bi = oi; }
        }
        if (l15 == 0) {
            int r = g + 8 * hi;
            codes_out[rowBase + r] = (float)bi;
            s_code[wave][r] = bi;
        }
    }
    __syncthreads();

    // epilogue: lane (l15,hi) handles row l15, dims [16*hi, 16*hi+16)
    int row  = rowBase + l15;
    int code = s_code[wave][l15];
    int ks   = hi * 16;
    float lp = 0.0f;
    for (int k = ks; k < ks + 16; ++k) {
        float r  = residual[(size_t)row * DIM + k];
        float e  = cbf[code * DIM + k];
        float rn = r - e;
        lp += rn * rn;
        residual[(size_t)row * DIM + k]          = rn;
        residual_h[(size_t)row * DIM + aswz(k)]  = (_Float16)rn;
    }
    for (int m = 16; m >= 1; m >>= 1) lp += __shfl_xor(lp, m, 32);
    if (lane == 0) atomicAdd(lossAcc, lp * LOSS_SCALE);
}

// ---------------------------------------------------------------------------
// Level-2 distance passes.  MODE 0: global d_min/d_max.  MODE 1: store
// s = -(d-mid)/amp as f16 (128 MB, L2-resident for the Sinkhorn loop).
template <int MODE>
__global__ __launch_bounds__(256) void rq_dist2_kernel(
    const _Float16* __restrict__ cbh, const _Float16* __restrict__ residual_h,
    _Float16* __restrict__ S,
    unsigned* __restrict__ dminO, unsigned* __restrict__ dmaxO)
{
    __shared__ __align__(32) _Float16 s_cb[KCB * DIM];
    __shared__ float s_c2[KCB];

    int tid = threadIdx.x;
    for (int i = tid; i < (KCB * DIM) / 8; i += 256)
        ((v8h*)s_cb)[i] = ((const v8h*)cbh)[i];
    __syncthreads();
    if (tid < KCB) {
        float s = 0.0f;
        for (int k = 0; k < DIM; ++k) {
            float x = (float)s_cb[tid * DIM + k];
            s += x * x;
        }
        s_c2[tid] = s;
    }
    __syncthreads();

    float mid = 0.0f, inv_amp = 0.0f;
    if (MODE == 1) {
        float dmin = funordered(*dminO);
        float dmax = funordered(*dmaxO);
        mid = 0.5f * (dmax + dmin);
        inv_amp = 1.0f / fmaxf(dmax - mid, 1e-5f);
    }

    int wave = tid >> 5, lane = tid & 31;
    int l15 = lane & 15, hi = lane >> 4;
    int rowBase = (blockIdx.x * 8 + wave) * 16;

    v16h a = *(const v16h*)(residual_h + (size_t)(rowBase + l15) * DIM + hi * 16);

    float r2p = 0.0f;
    for (int p = 0; p < 16; ++p) { float x = (float)a[p]; r2p += x * x; }
    r2p += __shfl_xor(r2p, 16, 32);
    float r2g[8];
    for (int g = 0; g < 8; ++g) r2g[g] = __shfl(r2p, g + 8 * hi, 32);

    float dmn = POS_INF, dmx = NEG_INF;

    for (int t = 0; t < 16; ++t) {
        const _Float16* bp = s_cb + (size_t)(t * 16 + l15) * DIM + hi * 8;
        v8h b0 = *(const v8h*)bp, b1 = *(const v8h*)(bp + 16);
        v16h b = __builtin_shufflevector(b0, b1, 0,1,2,3,4,5,6,7,8,9,10,11,12,13,14,15);
        v8f c = {};
        c = __builtin_amdgcn_wmma_f32_16x16x32_f16(false, a, false, b,
                                                   (short)0, c, false, false);
        int col = t * 16 + l15;
        float c2v = s_c2[col];
        for (int g = 0; g < 8; ++g) {
            float d = r2g[g] + c2v - 2.0f * c[g];
            if (MODE == 0) {
                dmn = fminf(dmn, d);
                dmx = fmaxf(dmx, d);
            } else {
                size_t row = (size_t)(rowBase + g + 8 * hi);
                S[row * KCB + col] = (_Float16)(-(d - mid) * inv_amp);
            }
        }
    }
    if (MODE == 0) {
        for (int m = 16; m >= 1; m >>= 1) {
            dmn = fminf(dmn, __shfl_xor(dmn, m, 32));
            dmx = fmaxf(dmx, __shfl_xor(dmx, m, 32));
        }
        if (lane == 0) {
            atomicMin(dminO, fordered(dmn));
            atomicMax(dmaxO, fordered(dmx));
        }
    }
}

// ---------------------------------------------------------------------------
// One fused Sinkhorn iteration (data pass): per row compute
// u_i = -logB - LSE_j(s_ij/eps + v_j), then fold u_i into per-column
// online-LSE partials for v.  512 WGs x 512 rows; wave = one row at a time,
// lane owns columns [8*lane, 8*lane+8).
__global__ __launch_bounds__(256) void rq_sinkhorn_rowcol(
    const _Float16* __restrict__ S, const float* __restrict__ v,
    float* __restrict__ gpart)
{
    __shared__ float s_v[KCB];
    __shared__ float s_pm[8 * KCB];
    __shared__ float s_ps[8 * KCB];

    int tid = threadIdx.x;
    if (tid < KCB) s_v[tid] = v[tid];
    __syncthreads();

    int wave = tid >> 5, lane = tid & 31;
    const float inv_eps = 1.0f / SK_EPS;

    float vloc[8];
    for (int k = 0; k < 8; ++k) vloc[k] = s_v[lane * 8 + k];

    float cm[8], cs[8];
    for (int k = 0; k < 8; ++k) { cm[k] = NEG_INF; cs[k] = 0.0f; }

    size_t rowBase = (size_t)blockIdx.x * 512 + (size_t)wave * 64;
    for (int r = 0; r < 64; ++r) {
        size_t row = rowBase + r;
        __builtin_prefetch(S + (row + 4) * KCB + lane * 8, 0, 0);
        v8h sv = *(const v8h*)(S + row * KCB + lane * 8);
        float x[8], m = NEG_INF;
        for (int k = 0; k < 8; ++k) {
            x[k] = (float)sv[k] * inv_eps + vloc[k];
            m = fmaxf(m, x[k]);
        }
        for (int mm = 16; mm >= 1; mm >>= 1)
            m = fmaxf(m, __shfl_xor(m, mm, 32));
        float s = 0.0f;
        for (int k = 0; k < 8; ++k) s += __expf(x[k] - m);
        for (int mm = 16; mm >= 1; mm >>= 1)
            s += __shfl_xor(s, mm, 32);
        float u = -LOG_B - (m + __logf(s));
        for (int k = 0; k < 8; ++k) {
            float y  = x[k] + u;
            float mn = fmaxf(cm[k], y);
            cs[k] = cs[k] * __expf(cm[k] - mn) + __expf(y - mn);
            cm[k] = mn;
        }
    }
    for (int k = 0; k < 8; ++k) {
        s_pm[wave * KCB + lane * 8 + k] = cm[k];
        s_ps[wave * KCB + lane * 8 + k] = cs[k];
    }
    __syncthreads();
    if (tid < KCB) {
        float m = NEG_INF;
        for (int w = 0; w < 8; ++w) m = fmaxf(m, s_pm[w * KCB + tid]);
        float s = 0.0f;
        for (int w = 0; w < 8; ++w)
            s += s_ps[w * KCB + tid] * __expf(s_pm[w * KCB + tid] - m);
        size_t o = ((size_t)blockIdx.x * KCB + tid) * 2;
        gpart[o]     = m;
        gpart[o + 1] = s;
    }
}

// Finalize v_j = -logK - LSE over the 512 per-WG partials.
__global__ __launch_bounds__(256) void rq_sinkhorn_vfinal(
    const float* __restrict__ gpart, float* __restrict__ v,
    unsigned* __restrict__ bad)
{
    int j = threadIdx.x;
    float m = NEG_INF, s = 0.0f;
    for (int wg = 0; wg < 512; ++wg) {
        size_t o = ((size_t)wg * KCB + j) * 2;
        float pm = gpart[o], ps = gpart[o + 1];
        float mn = fmaxf(m, pm);
        s = s * __expf(m - mn) + ps * __expf(pm - mn);
        m = mn;
    }
    float vj = -LOG_K - (m + __logf(s));
    if (!__builtin_isfinite(vj)) atomicOr(bad, 1u);
    v[j] = vj;
}

// ---------------------------------------------------------------------------
// Level-2 finalize: code2 = argmax_j(s/eps + v_j) (u is per-row constant;
// bad -> drop v, which equals the argmin-dists fallback).  Then
// quantized_total = z - (residual2 - e2) and the level-2 loss.
__global__ __launch_bounds__(256) void rq_final_kernel(
    const _Float16* __restrict__ S, const float* __restrict__ v,
    const unsigned* __restrict__ bad,
    const float* __restrict__ cb2f, const float* __restrict__ residual,
    const float* __restrict__ z,
    float* __restrict__ out_codes2, float* __restrict__ out_qt,
    float* __restrict__ lossAcc)
{
    __shared__ float s_v[KCB];
    int tid = threadIdx.x;
    unsigned isbad = *bad;
    if (tid < KCB) s_v[tid] = isbad ? 0.0f : v[tid];
    __syncthreads();

    int wave = tid >> 5, lane = tid & 31;
    const float inv_eps = 1.0f / SK_EPS;
    float vloc[8];
    for (int k = 0; k < 8; ++k) vloc[k] = s_v[lane * 8 + k];

    float lp = 0.0f;
    size_t rowBase = (size_t)blockIdx.x * 256 + (size_t)wave * 32;
    for (int r = 0; r < 32; ++r) {
        size_t row = rowBase + r;
        v8h sv = *(const v8h*)(S + row * KCB + lane * 8);
        float bv = NEG_INF; int bi = 0;
        for (int k = 0; k < 8; ++k) {
            float x = (float)sv[k] * inv_eps + vloc[k];
            int col = lane * 8 + k;
            if (x > bv) { bv = x; bi = col; }
        }
        for (int m = 16; m >= 1; m >>= 1) {
            float ov = __shfl_xor(bv, m, 32);
            int   oi = __shfl_xor(bi, m, 32);
            if (ov > bv || (ov == bv && oi < bi)) { bv = ov; bi = oi; }
        }
        // all lanes agree on bi; lane == dim index (DIM == warp width == 32)
        float rr = residual[row * DIM + lane];
        float e  = cb2f[bi * DIM + lane];
        float zz = z[row * DIM + lane];
        float rn = rr - e;
        lp += rn * rn;
        out_qt[row * DIM + lane] = zz - rn;
        if (lane == 0) out_codes2[row] = (float)bi;
    }
    for (int m = 16; m >= 1; m >>= 1) lp += __shfl_xor(lp, m, 32);
    if (lane == 0) atomicAdd(lossAcc, lp * LOSS_SCALE);
}

__global__ void rq_writeloss_kernel(const float* __restrict__ lossAcc,
                                    float* __restrict__ out)
{
    if (threadIdx.x == 0 && blockIdx.x == 0) out[0] = *lossAcc;
}

// ---------------------------------------------------------------------------
extern "C" void kernel_launch(void* const* d_in, const int* in_sizes, int n_in,
                              void* d_out, int out_size, void* d_ws,
                              size_t ws_size, hipStream_t stream)
{
    const float* z   = (const float*)d_in[0];
    const float* cb0 = (const float*)d_in[1];
    const float* cb1 = (const float*)d_in[2];
    const float* cb2 = (const float*)d_in[3];

    float* out       = (float*)d_out;
    float* out_c0    = out;
    float* out_c1    = out + B_SIZE;
    float* out_c2    = out + 2 * (size_t)B_SIZE;
    float* out_qt    = out + 3 * (size_t)B_SIZE;
    float* out_loss  = out + 3 * (size_t)B_SIZE + (size_t)B_SIZE * DIM;

    // workspace carve-up (256B-aligned)
    size_t off = 0;
    auto carve = [&](size_t bytes) {
        void* p = (char*)d_ws + off;
        off += (bytes + 255) & ~(size_t)255;
        return p;
    };
    float*     residual   = (float*)    carve((size_t)B_SIZE * DIM * 4);
    _Float16*  residual_h = (_Float16*) carve((size_t)B_SIZE * DIM * 2);
    _Float16*  Smat       = (_Float16*) carve((size_t)B_SIZE * KCB * 2);
    _Float16*  cbh0       = (_Float16*) carve((size_t)KCB * DIM * 2);
    _Float16*  cbh1       = (_Float16*) carve((size_t)KCB * DIM * 2);
    _Float16*  cbh2       = (_Float16*) carve((size_t)KCB * DIM * 2);
    float*     vvec       = (float*)    carve((size_t)KCB * 4);
    float*     gpart      = (float*)    carve((size_t)512 * KCB * 2 * 4);
    float*     lossAcc    = (float*)    carve(4);
    unsigned*  dminO      = (unsigned*) carve(4);
    unsigned*  dmaxO      = (unsigned*) carve(4);
    unsigned*  badflag    = (unsigned*) carve(4);

    rq_prep_kernel<<<(B_SIZE * DIM) / 256, 256, 0, stream>>>(
        z, cb0, cb1, cb2, residual, residual_h, cbh0, cbh1, cbh2,
        vvec, lossAcc, dminO, dmaxO, badflag);

    // levels 0 and 1: argmin quantization + residual update
    rq_level_kernel<<<B_SIZE / 128, 256, 0, stream>>>(
        cbh0, cb0, residual, residual_h, out_c0, lossAcc);
    rq_level_kernel<<<B_SIZE / 128, 256, 0, stream>>>(
        cbh1, cb1, residual, residual_h, out_c1, lossAcc);

    // level 2: global d-range, then L2-resident f16 score matrix
    rq_dist2_kernel<0><<<B_SIZE / 128, 256, 0, stream>>>(
        cbh2, residual_h, Smat, dminO, dmaxO);
    rq_dist2_kernel<1><<<B_SIZE / 128, 256, 0, stream>>>(
        cbh2, residual_h, Smat, dminO, dmaxO);

    // 50 Sinkhorn iterations in (u,v) vector form: one data pass each
    for (int it = 0; it < SK_ITERS; ++it) {
        rq_sinkhorn_rowcol<<<512, 256, 0, stream>>>(Smat, vvec, gpart);
        rq_sinkhorn_vfinal<<<1, 256, 0, stream>>>(gpart, vvec, badflag);
    }

    rq_final_kernel<<<B_SIZE / 256, 256, 0, stream>>>(
        Smat, vvec, badflag, cb2, residual, z, out_c2, out_qt, lossAcc);

    rq_writeloss_kernel<<<1, 64, 0, stream>>>(lossAcc, out_loss);
}
